// HANLayer_21492016349917
// MI455X (gfx1250) — matrix operand order repaired and checked by reading, not compiled, not run on hardware
//
#include <hip/hip_runtime.h>
#include <stdint.h>
#include <math.h>

#define N_A_    100000
#define N_B_    100000
#define IN_D    256
#define OUT_D   256
#define N_E     320000
#define LN_EPS_ 1e-5f

// ---- plain clang ext_vector types (POD, union-safe; HIP's uint4 has ctors) ----
typedef __bf16        v16bf  __attribute__((ext_vector_type(16)));
typedef float         v8f    __attribute__((ext_vector_type(8)));
typedef unsigned int  uint4n __attribute__((ext_vector_type(4)));
typedef unsigned int  uint2n __attribute__((ext_vector_type(2)));
typedef float         flt4n  __attribute__((ext_vector_type(4)));

__device__ __forceinline__ unsigned short f2bf(float f) {
    unsigned int u = __float_as_uint(f);
    u += 0x7FFFu + ((u >> 16) & 1u);          // round-to-nearest-even
    return (unsigned short)(u >> 16);
}
__device__ __forceinline__ unsigned int pack2bf(float lo, float hi) {
    return (unsigned int)f2bf(lo) | ((unsigned int)f2bf(hi) << 16);
}
__device__ __forceinline__ float bflo(unsigned int u) { return __uint_as_float(u << 16); }
__device__ __forceinline__ float bfhi(unsigned int u) { return __uint_as_float(u & 0xFFFF0000u); }

__device__ __forceinline__ float waveSum(float v) {
    #pragma unroll
    for (int off = 16; off > 0; off >>= 1) v += __shfl_xor(v, off, 32);
    return v;
}

// ---------------------------------------------------------------------------
// 1) Convert x_B (f32, row-major) -> bf16 once.  4 elems / thread.
// ---------------------------------------------------------------------------
__global__ __launch_bounds__(256)
void cvt_f32_bf16(const float* __restrict__ x, unsigned int* __restrict__ out, int n4) {
    int i = blockIdx.x * 256 + threadIdx.x;
    if (i < n4) {
        flt4n f = ((const flt4n*)x)[i];
        uint2n o;
        o.x = pack2bf(f.x, f.y);
        o.y = pack2bf(f.z, f.w);
        ((uint2n*)out)[i] = o;
    }
}

// ---------------------------------------------------------------------------
// 2) Pre-pack W0/W1 into WMMA B-fragment order (ISA 7.12.2, 16-bit B 32x16):
//    lane 0-15 -> col N=lane, K=0..7,16..23 (VGPR v holds K=2v+8*(v>>2), +1);
//    lane 16-31 -> col N=lane-16, K=8..15,24..31 (same, +8).
//    dword index (per path) = ((kt*16 + nb)*32 + lane)*8 + v
// ---------------------------------------------------------------------------
__global__ __launch_bounds__(256)
void pack_weights(const float* __restrict__ W0, const float* __restrict__ W1,
                  unsigned int* __restrict__ wp0, unsigned int* __restrict__ wp1) {
    int idx  = blockIdx.x * 256 + threadIdx.x;      // 0..65535  (bit15 = path)
    int v    = idx & 7;
    int lane = (idx >> 3) & 31;
    int nb   = (idx >> 8) & 15;
    int kt   = (idx >> 12) & 7;
    int p    = (idx >> 15) & 1;
    int n  = nb * 16 + (lane & 15);
    int k0 = kt * 32 + (lane >> 4) * 8 + 2 * v + (v >> 2) * 8;
    const float*  W  = p ? W1 : W0;
    unsigned int* wp = p ? wp1 : wp0;
    wp[idx & 0x7FFF] = pack2bf(W[(size_t)k0 * OUT_D + n], W[(size_t)(k0 + 1) * OUT_D + n]);
}

// ---------------------------------------------------------------------------
// 3) Dual projection GEMM: proj_p = bf16(x_B) @ bf16(W_p) + b_p, stored bf16.
//    One wave per (16-row tile, 16-col block, path); K=256 -> 8 WMMA/wave.
//    A frag (16-bit A 16x32): lane m=lane&15 is row; h=lane>>4 selects
//    K∈{h*8..h*8+7} ∪ {16+h*8..16+h*8+7}? -> elements 0..7 = K=kb..kb+7,
//    elements 8..15 = K=kb+16..kb+23, kb = kt*32 + h*8.  Contiguous 16B loads.
// ---------------------------------------------------------------------------
__global__ __launch_bounds__(256)
void proj_gemm(const unsigned short* __restrict__ xbf,
               const unsigned int*  __restrict__ wp0,
               const unsigned int*  __restrict__ wp1,
               const float* __restrict__ bias0, const float* __restrict__ bias1,
               unsigned short* __restrict__ proj0, unsigned short* __restrict__ proj1) {
    const int lane = threadIdx.x & 31;
    const int wid  = blockIdx.x * 8 + (threadIdx.x >> 5);   // 0..199999
    const int tile = wid >> 5;                              // 6250 row tiles
    const int rem  = wid & 31;
    const int p    = rem >> 4;                              // path 0/1
    const int nb   = rem & 15;                              // 16-col block
    const int m    = lane & 15;
    const int h    = lane >> 4;
    const int row0 = tile * 16;

    const unsigned short* arow = xbf + (size_t)(row0 + m) * IN_D;
    const unsigned int*   wp   = p ? wp1 : wp0;
    const float*          bias = p ? bias1 : bias0;
    unsigned short*       proj = p ? proj1 : proj0;

    v8f c = {0.f, 0.f, 0.f, 0.f, 0.f, 0.f, 0.f, 0.f};
    #pragma unroll
    for (int kt = 0; kt < 8; ++kt) {
        const int kb = kt * 32 + h * 8;
        union { uint4n q[2]; v16bf v; } a, b;
        a.q[0] = *(const uint4n*)(arow + kb);
        a.q[1] = *(const uint4n*)(arow + kb + 16);
        const uint4n* bp = (const uint4n*)(wp + (size_t)(((kt * 16 + nb) * 32 + lane) * 8));
        b.q[0] = bp[0];
        b.q[1] = bp[1];
        // D = A x B + C,  f32 accumulate
        c = __builtin_amdgcn_wmma_f32_16x16x32_bf16(false, a.v, false, b.v,
                                                    (short)0, c, false, false);
    }
    // C/D layout: lane&15 = column, VGPR v = row v + 8*(lane>>4)
    const int   col = nb * 16 + m;
    const float bv  = bias[col];
    #pragma unroll
    for (int v = 0; v < 8; ++v) {
        proj[(size_t)(row0 + v + 8 * h) * OUT_D + col] = f2bf(c[v] + bv);
    }
}

// ---------------------------------------------------------------------------
// 4) CSR build: histogram -> exclusive scan -> fill
// ---------------------------------------------------------------------------
__global__ __launch_bounds__(256)
void hist_kernel(const int* __restrict__ ei0, const int* __restrict__ ei1,
                 int* __restrict__ deg0, int* __restrict__ deg1) {
    int e = blockIdx.x * 256 + threadIdx.x;
    if (e < N_E) {
        atomicAdd(&deg0[ei0[N_E + e]], 1);   // ei[1] = dst
        atomicAdd(&deg1[ei1[N_E + e]], 1);
    }
}

__global__ __launch_bounds__(1024)
void scan_kernel(const int* __restrict__ deg0, const int* __restrict__ deg1,
                 int* __restrict__ rs0, int* __restrict__ rs1,
                 int* __restrict__ cur0, int* __restrict__ cur1) {
    const int* deg = blockIdx.x ? deg1 : deg0;
    int*       rs  = blockIdx.x ? rs1  : rs0;
    int*       cur = blockIdx.x ? cur1 : cur0;

    __shared__ int sm[1024];
    __shared__ int s_running;
    if (threadIdx.x == 0) s_running = 0;
    __syncthreads();

    for (int base = 0; base < N_A_; base += 1024) {
        int i = base + (int)threadIdx.x;
        int v = (i < N_A_) ? deg[i] : 0;
        sm[threadIdx.x] = v;
        __syncthreads();
        int acc = v;
        for (int offd = 1; offd < 1024; offd <<= 1) {
            int t = (threadIdx.x >= (unsigned)offd) ? sm[threadIdx.x - offd] : 0;
            __syncthreads();
            acc += t;
            sm[threadIdx.x] = acc;
            __syncthreads();
        }
        int run  = s_running;
        int excl = acc - v;
        if (i < N_A_) { rs[i] = run + excl; cur[i] = run + excl; }
        int total = sm[1023];
        __syncthreads();
        if (threadIdx.x == 0) s_running = run + total;
        __syncthreads();
    }
    if (threadIdx.x == 0) rs[N_A_] = s_running;
}

__global__ __launch_bounds__(256)
void fill_kernel(const int* __restrict__ ei0, const int* __restrict__ ei1,
                 int* __restrict__ cur0, int* __restrict__ cur1,
                 int* __restrict__ csr0, int* __restrict__ csr1) {
    int e = blockIdx.x * 256 + threadIdx.x;
    if (e < N_E) {
        int d0 = ei0[N_E + e];
        int p0 = atomicAdd(&cur0[d0], 1);
        csr0[p0] = ei0[e];                   // source node
        int d1 = ei1[N_E + e];
        int p1 = atomicAdd(&cur1[d1], 1);
        csr1[p1] = ei1[e];
    }
}

// ---------------------------------------------------------------------------
// 5) Fused: per-dst mean-aggregate (both paths) + semantic attention softmax
//    + ReLU + LayerNorm.  One wave32 per destination node; 8 cols/lane.
// ---------------------------------------------------------------------------
__global__ __launch_bounds__(256)
void agg_fuse(const unsigned short* __restrict__ proj0,
              const unsigned short* __restrict__ proj1,
              const int* __restrict__ rs0, const int* __restrict__ csr0,
              const int* __restrict__ rs1, const int* __restrict__ csr1,
              const float* __restrict__ sem, const float* __restrict__ gam,
              const float* __restrict__ bet, float* __restrict__ outA) {
    const int node = blockIdx.x * 8 + (threadIdx.x >> 5);
    if (node >= N_A_) return;
    const int lane  = threadIdx.x & 31;
    const int cbase = lane * 8;

    float a0[8] = {0,0,0,0,0,0,0,0};
    float a1[8] = {0,0,0,0,0,0,0,0};

    const int s0 = rs0[node], e0 = rs0[node + 1];
    for (int i = s0; i < e0; ++i) {
        const int src = csr0[i];
        const uint4n u = *((const uint4n*)(proj0 + (size_t)src * OUT_D) + lane);
        a0[0] += bflo(u.x); a0[1] += bfhi(u.x); a0[2] += bflo(u.y); a0[3] += bfhi(u.y);
        a0[4] += bflo(u.z); a0[5] += bfhi(u.z); a0[6] += bflo(u.w); a0[7] += bfhi(u.w);
    }
    const int s1 = rs1[node], e1 = rs1[node + 1];
    for (int i = s1; i < e1; ++i) {
        const int src = csr1[i];
        const uint4n u = *((const uint4n*)(proj1 + (size_t)src * OUT_D) + lane);
        a1[0] += bflo(u.x); a1[1] += bfhi(u.x); a1[2] += bflo(u.y); a1[3] += bfhi(u.y);
        a1[4] += bflo(u.z); a1[5] += bfhi(u.z); a1[6] += bflo(u.w); a1[7] += bfhi(u.w);
    }

    const float inv0 = 1.0f / fmaxf((float)(e0 - s0), 1.0f);
    const float inv1 = 1.0f / fmaxf((float)(e1 - s1), 1.0f);
    float sc0 = 0.f, sc1 = 0.f;
    #pragma unroll
    for (int j = 0; j < 8; ++j) {
        a0[j] *= inv0;
        a1[j] *= inv1;
        const float sv = sem[cbase + j];
        sc0 += tanhf(a0[j]) * sv;
        sc1 += tanhf(a1[j]) * sv;
    }
    sc0 = waveSum(sc0);
    sc1 = waveSum(sc1);

    // softmax over the 2 meta-paths
    const float mx = fmaxf(sc0, sc1);
    float w0 = expf(sc0 - mx), w1 = expf(sc1 - mx);
    const float winv = 1.0f / (w0 + w1);
    w0 *= winv; w1 *= winv;

    float fv[8];
    float fsum = 0.f, fsq = 0.f;
    #pragma unroll
    for (int j = 0; j < 8; ++j) {
        float f = fmaxf(w0 * a0[j] + w1 * a1[j], 0.0f);   // ReLU(fused)
        fv[j] = f; fsum += f; fsq += f * f;
    }
    const float mu   = waveSum(fsum) * (1.0f / OUT_D);
    const float var  = waveSum(fsq)  * (1.0f / OUT_D) - mu * mu;
    const float rstd = rsqrtf(var + LN_EPS_);

    float* orow = outA + (size_t)node * OUT_D + cbase;
    flt4n o0, o1;
    o0.x = (fv[0] - mu) * rstd * gam[cbase + 0] + bet[cbase + 0];
    o0.y = (fv[1] - mu) * rstd * gam[cbase + 1] + bet[cbase + 1];
    o0.z = (fv[2] - mu) * rstd * gam[cbase + 2] + bet[cbase + 2];
    o0.w = (fv[3] - mu) * rstd * gam[cbase + 3] + bet[cbase + 3];
    o1.x = (fv[4] - mu) * rstd * gam[cbase + 4] + bet[cbase + 4];
    o1.y = (fv[5] - mu) * rstd * gam[cbase + 5] + bet[cbase + 5];
    o1.z = (fv[6] - mu) * rstd * gam[cbase + 6] + bet[cbase + 6];
    o1.w = (fv[7] - mu) * rstd * gam[cbase + 7] + bet[cbase + 7];
    ((flt4n*)orow)[0] = o0;
    ((flt4n*)orow)[1] = o1;
}

// ---------------------------------------------------------------------------
extern "C" void kernel_launch(void* const* d_in, const int* in_sizes, int n_in,
                              void* d_out, int out_size, void* d_ws, size_t ws_size,
                              hipStream_t stream) {
    (void)in_sizes; (void)n_in; (void)out_size; (void)ws_size;
    // inputs (setup_inputs order): x_A, x_B, ei0, ei1, W0, b0, W1, b1, sem, gamma, beta
    const float* xB  = (const float*)d_in[1];
    const int*   ei0 = (const int*)d_in[2];
    const int*   ei1 = (const int*)d_in[3];
    const float* W0  = (const float*)d_in[4];
    const float* b0  = (const float*)d_in[5];
    const float* W1  = (const float*)d_in[6];
    const float* b1  = (const float*)d_in[7];
    const float* sem = (const float*)d_in[8];
    const float* gam = (const float*)d_in[9];
    const float* bet = (const float*)d_in[10];

    float* outA = (float*)d_out;
    float* outB = outA + (size_t)N_A_ * OUT_D;

    uint8_t* base = (uint8_t*)d_ws;
    size_t off = 0;
    auto take = [&](size_t bytes) -> void* {
        void* p = base + off;
        off += (bytes + 255) & ~(size_t)255;
        return p;
    };
    unsigned short* xbf   = (unsigned short*)take((size_t)N_B_ * IN_D * 2);   // 51.2 MB
    unsigned short* proj0 = (unsigned short*)take((size_t)N_B_ * OUT_D * 2);  // 51.2 MB
    unsigned short* proj1 = (unsigned short*)take((size_t)N_B_ * OUT_D * 2);  // 51.2 MB
    unsigned int*   wp0   = (unsigned int*)take(8 * 16 * 32 * 8 * 4);         // 128 KB
    unsigned int*   wp1   = (unsigned int*)take(8 * 16 * 32 * 8 * 4);
    int* deg0 = (int*)take((size_t)2 * N_A_ * 4);                             // deg0|deg1
    int* deg1 = deg0 + N_A_;
    int* rs0  = (int*)take((size_t)(N_A_ + 1) * 4);
    int* rs1  = (int*)take((size_t)(N_A_ + 1) * 4);
    int* cur0 = (int*)take((size_t)N_A_ * 4);
    int* cur1 = (int*)take((size_t)N_A_ * 4);
    int* csr0 = (int*)take((size_t)N_E * 4);
    int* csr1 = (int*)take((size_t)N_E * 4);

    hipMemsetAsync(deg0, 0, (size_t)2 * N_A_ * sizeof(int), stream);
    hipMemsetAsync(outB, 0, (size_t)N_B_ * OUT_D * sizeof(float), stream);   // out_B = zeros

    cvt_f32_bf16<<<(N_B_ * IN_D / 4 + 255) / 256, 256, 0, stream>>>(
        xB, (unsigned int*)xbf, N_B_ * IN_D / 4);
    pack_weights<<<65536 / 256, 256, 0, stream>>>(W0, W1, wp0, wp1);

    // 6250 row tiles * 16 col blocks * 2 paths = 200000 waves, 8 waves/block
    proj_gemm<<<200000 / 8, 256, 0, stream>>>(xbf, wp0, wp1, b0, b1, proj0, proj1);

    hist_kernel<<<(N_E + 255) / 256, 256, 0, stream>>>(ei0, ei1, deg0, deg1);
    scan_kernel<<<2, 1024, 0, stream>>>(deg0, deg1, rs0, rs1, cur0, cur1);
    fill_kernel<<<(N_E + 255) / 256, 256, 0, stream>>>(ei0, ei1, cur0, cur1, csr0, csr1);

    // 100000 dst nodes, one wave each, 8 waves/block
    agg_fuse<<<N_A_ / 8, 256, 0, stream>>>(proj0, proj1, rs0, csr0, rs1, csr1,
                                           sem, gam, bet, outA);
}
